// Detection_53420803228400
// MI455X (gfx1250) — compile-verified, matrix operand order checked by reading, not run on hardware
//
#include <hip/hip_runtime.h>
#include <cstdint>
#include <cstddef>

// ---------------- problem constants ----------------
constexpr int   NB        = 16;        // batch
constexpr int   NA        = 262144;    // anchors
constexpr int   NC        = 5;         // classes incl background
constexpr int   CMX       = 4;         // foreground classes
constexpr int   TOPK      = 200;
constexpr float F_CLS_THR = 0.01f;
constexpr float F_OVERLAP = 0.45f;
constexpr float F_VAR_C   = 0.1f;
constexpr float F_VAR_W   = 0.2f;

// radix-select binning: positive-float bits are order-preserving.
// bits >> 20 keeps sign+exp+3 mantissa bits; scores in (0,1] -> bin < 1024.
constexpr int NBINS     = 1024;
constexpr int BIN_SHIFT = 20;
constexpr int CAP       = 2048;       // candidate list capacity per (b,c)

constexpr int THREADS      = 256;
constexpr int TILE         = 1024;            // anchors staged per LDS tile
constexpr int TPB          = 8;               // tiles per block
constexpr int CHUNK        = TILE * TPB;      // 8192 anchors per block
constexpr int CHUNKS_PER_B = NA / CHUNK;      // 32

// NMS tiling: pad 200 boxes to 208 = 13 x 16 for WMMA tiles
constexpr int NTILE = 13;
constexpr int NT    = NTILE * 16;     // 208

// WMMA vector types (CDNA5 wave32)
typedef __attribute__((ext_vector_type(16))) _Float16 v16h;
typedef __attribute__((ext_vector_type(8)))  float    v8f;

// ---------------- CDNA5 async global->LDS helpers ----------------
// Each tile = TILE*NC*4 = 20480 bytes = 256 threads * 5 * b128.
__device__ __forceinline__ void async_tile_copy(const float* __restrict__ gsrc,
                                                float* lbuf, int tid) {
  unsigned int lds = (unsigned int)reinterpret_cast<uintptr_t>(lbuf) + (unsigned)tid * 16u;
  unsigned long long g = (unsigned long long)reinterpret_cast<uintptr_t>(gsrc)
                         + (unsigned long long)tid * 16ull;
#pragma unroll
  for (int i = 0; i < 5; ++i) {
    asm volatile("global_load_async_to_lds_b128 %0, %1, off"
                 :: "v"(lds), "v"(g) : "memory");
    lds += 256u * 16u;
    g   += 256ull * 16ull;
  }
}
#define WAIT_ASYNC(n) asm volatile("s_wait_asynccnt " #n ::: "memory")

// ---------------- small math helpers ----------------
__device__ __forceinline__ void softmax5(const float x[5], float p[5]) {
  float m = fmaxf(fmaxf(fmaxf(x[0], x[1]), fmaxf(x[2], x[3])), x[4]);
  float s = 0.f;
#pragma unroll
  for (int i = 0; i < 5; ++i) { p[i] = __expf(x[i] - m); s += p[i]; }
  float inv = 1.0f / s;
#pragma unroll
  for (int i = 0; i < 5; ++i) p[i] *= inv;
}

// ---------------- kernel 0: workspace reset ----------------
__global__ void zero_ws_kernel(unsigned int* __restrict__ p, int nwords) {
  int i = blockIdx.x * blockDim.x + threadIdx.x;
  if (i < nwords) p[i] = 0u;
}

// ---------------- kernel 1: stream + per-(b,c) score histogram ----------------
__global__ void __launch_bounds__(THREADS)
scan_hist_kernel(const float* __restrict__ cls, unsigned int* __restrict__ hist) {
  __shared__ unsigned int sh_hist[CMX * NBINS];   // 16 KB
  __shared__ float sbuf[2][TILE * NC];            // 40 KB, double buffered
  const int tid   = threadIdx.x;
  const int b     = blockIdx.x / CHUNKS_PER_B;
  const int chunk = blockIdx.x % CHUNKS_PER_B;
  const int a0    = chunk * CHUNK;

  for (int i = tid; i < CMX * NBINS; i += THREADS) sh_hist[i] = 0u;

  const float* gb = cls + ((size_t)b * NA + a0) * NC;
  async_tile_copy(gb, sbuf[0], tid);
  for (int t = 0; t < TPB; ++t) {
    const int cur = t & 1;
    if (t + 1 < TPB) {
      async_tile_copy(gb + (size_t)(t + 1) * TILE * NC, sbuf[cur ^ 1], tid);
      WAIT_ASYNC(5);   // tile t complete, tile t+1 still in flight
    } else {
      WAIT_ASYNC(0);
    }
    __syncthreads();
    for (int a = tid; a < TILE; a += THREADS) {
      float x[5], p[5];
#pragma unroll
      for (int q = 0; q < 5; ++q) x[q] = sbuf[cur][a * NC + q];
      softmax5(x, p);
#pragma unroll
      for (int c = 1; c < 5; ++c) {
        if (p[c] > F_CLS_THR) {
          unsigned bin = __float_as_uint(p[c]) >> BIN_SHIFT;
          atomicAdd(&sh_hist[(c - 1) * NBINS + (int)bin], 1u);
        }
      }
    }
    __syncthreads();
  }
  unsigned int* gh = hist + (size_t)b * CMX * NBINS;
  for (int i = tid; i < CMX * NBINS; i += THREADS) {
    unsigned v = sh_hist[i];
    if (v) atomicAdd(&gh[i], v);
  }
}

// ---------------- kernel 2: find per-(b,c) threshold bin ----------------
__global__ void select_thr_kernel(const unsigned int* __restrict__ hist,
                                  unsigned int* __restrict__ thrBin) {
  int bc = blockIdx.x * blockDim.x + threadIdx.x;
  if (bc >= NB * CMX) return;
  const unsigned int* h = hist + (size_t)bc * NBINS;
  unsigned int cum = 0; int t = 0;
  for (int bin = NBINS - 1; bin >= 0; --bin) {
    cum += h[bin];
    if (cum >= (unsigned)TOPK) { t = bin; break; }
  }
  thrBin[bc] = (unsigned int)t;
}

// ---------------- kernel 3: re-stream and compact candidates ----------------
__global__ void __launch_bounds__(THREADS)
compact_kernel(const float* __restrict__ cls, const unsigned int* __restrict__ thrBin,
               unsigned int* __restrict__ candCnt, float* __restrict__ candScore,
               unsigned int* __restrict__ candIdx) {
  __shared__ float sbuf[2][TILE * NC];
  const int tid   = threadIdx.x;
  const int b     = blockIdx.x / CHUNKS_PER_B;
  const int chunk = blockIdx.x % CHUNKS_PER_B;
  const int a0    = chunk * CHUNK;
  unsigned thr[CMX];
#pragma unroll
  for (int c = 0; c < CMX; ++c) thr[c] = thrBin[b * CMX + c];

  const float* gb = cls + ((size_t)b * NA + a0) * NC;
  async_tile_copy(gb, sbuf[0], tid);
  for (int t = 0; t < TPB; ++t) {
    const int cur = t & 1;
    if (t + 1 < TPB) {
      async_tile_copy(gb + (size_t)(t + 1) * TILE * NC, sbuf[cur ^ 1], tid);
      WAIT_ASYNC(5);
    } else {
      WAIT_ASYNC(0);
    }
    __syncthreads();
    for (int a = tid; a < TILE; a += THREADS) {
      float x[5], p[5];
#pragma unroll
      for (int q = 0; q < 5; ++q) x[q] = sbuf[cur][a * NC + q];
      softmax5(x, p);
      const int n = a0 + t * TILE + a;
#pragma unroll
      for (int c = 1; c < 5; ++c) {
        float pc = p[c];
        if (pc > F_CLS_THR && (__float_as_uint(pc) >> BIN_SHIFT) >= thr[c - 1]) {
          const int bc = b * CMX + (c - 1);
          unsigned pos = atomicAdd(&candCnt[bc], 1u);
          if (pos < (unsigned)CAP) {
            candScore[(size_t)bc * CAP + pos] = pc;
            candIdx[(size_t)bc * CAP + pos]   = (unsigned)n;
          }
        }
      }
    }
    __syncthreads();
  }
}

// ---------------- kernel 4: sort, gather, WMMA-tiled suppression matrix, NMS ---
__global__ void __launch_bounds__(THREADS)
topk_nms_kernel(const float* __restrict__ loc, const float* __restrict__ cls,
                const float* __restrict__ defs,
                const float* __restrict__ candScore, const unsigned int* __restrict__ candIdx,
                const unsigned int* __restrict__ candCnt, float* __restrict__ out) {
  __shared__ float s_score[CAP];
  __shared__ int   s_idx[CAP];
  __shared__ float s_l[NT];
  __shared__ float s_r[NT];
  __shared__ unsigned char s_valid[TOPK];
  __shared__ unsigned int s_sup[NT][7];   // suppression bit-matrix (iou > thr)
  __shared__ unsigned int s_keepw[7];

  const int bc  = blockIdx.x;
  const int b   = bc / CMX;
  const int tid = threadIdx.x;
  const int nc  = (int)min(candCnt[bc], (unsigned)CAP);

  for (int i = tid; i < CAP; i += THREADS) {
    if (i < nc) {
      s_score[i] = candScore[(size_t)bc * CAP + i];
      s_idx[i]   = (int)candIdx[(size_t)bc * CAP + i];
    } else {
      s_score[i] = -1.0f;
      s_idx[i]   = 0x7FFFFFFF;
    }
  }
  for (int i = tid; i < NT * 7; i += THREADS) (&s_sup[0][0])[i] = 0u;
  __syncthreads();

  // bitonic sort: descending score, ties by ascending index
  for (int kk = 2; kk <= CAP; kk <<= 1) {
    for (int j = kk >> 1; j > 0; j >>= 1) {
      for (int t = tid; t < CAP / 2; t += THREADS) {
        int i   = ((t & ~(j - 1)) << 1) | (t & (j - 1));
        int ixj = i + j;
        float si = s_score[i], sj = s_score[ixj];
        int   ii = s_idx[i],   ij = s_idx[ixj];
        bool better_i = (si > sj) || (si == sj && ii < ij);
        bool dir = ((i & kk) == 0);          // final pass: descending everywhere
        if (dir != better_i) {
          s_score[i] = sj; s_score[ixj] = si;
          s_idx[i]   = ij; s_idx[ixj]   = ii;
        }
      }
      __syncthreads();
    }
  }

  // output pointers (flat concat in return order, all as f32)
  float* out_boxes  = out;                                   // [NB][CMX][TOPK][2]
  float* out_scores = out_boxes + (size_t)NB * CMX * TOPK * 2;
  float* out_cls    = out_scores + (size_t)NB * CMX * TOPK;  // [..][TOPK][NC]
  float* out_keep   = out_cls + (size_t)NB * CMX * TOPK * NC;
  float* out_idx    = out_keep + (size_t)NB * CMX * TOPK;

  for (int j = tid; j < TOPK; j += THREADS) {
    float s = s_score[j];
    bool valid = (s > F_CLS_THR);
    int n = valid ? s_idx[j] : 0;           // safe gather for padded slots
    float lx = loc[((size_t)b * NA + n) * 2 + 0];
    float lw = loc[((size_t)b * NA + n) * 2 + 1];
    float dc = defs[(size_t)n * 2 + 0];
    float dw = defs[(size_t)n * 2 + 1];
    float cc = dc + lx * F_VAR_C * dw;
    float ww = dw * __expf(lw * F_VAR_W);
    float l = cc - 0.5f * ww, r = cc + 0.5f * ww;
    size_t ob = (size_t)bc * TOPK + j;
    out_boxes[ob * 2 + 0] = l;
    out_boxes[ob * 2 + 1] = r;
    out_scores[ob] = valid ? s : 0.0f;
    out_idx[ob]    = (float)n;
    float x[5], p[5];
#pragma unroll
    for (int q = 0; q < 5; ++q) x[q] = cls[((size_t)b * NA + n) * NC + q];
    softmax5(x, p);
#pragma unroll
    for (int q = 0; q < 5; ++q) out_cls[ob * NC + q] = p[q];
    s_l[j] = l; s_r[j] = r; s_valid[j] = valid ? 1 : 0;
  }
  for (int j = TOPK + tid; j < NT; j += THREADS) { s_l[j] = 0.0f; s_r[j] = 0.0f; }
  __syncthreads();

  // Build the 200x200 suppression bit-matrix, one 16x16 pair-tile per wave.
  // union base term len_i + len_j is a rank-2 outer sum -> V_WMMA_F32_16X16X32_F16:
  //   A[m][k] = {len_i(m), 1, 0..}   B[k][n] = {1; len_j(n); 0..}
  //   D[i][j] = len_i + len_j   (f32 accumulate)
  // iou(i,j) > thr  <=>  inter*(1+thr) > thr*(len_i + len_j)
  const int wave = tid >> 5;
  const int lane = tid & 31;
  for (int tile = wave; tile < NTILE * NTILE; tile += THREADS / 32) {
    const int ti = tile / NTILE, tj = tile % NTILE;
    const int i0 = ti * 16, j0 = tj * 16;
    v16h Am = {};
    v16h Bm = {};
    if (lane < 16) {
      float len_i = s_r[i0 + lane] - s_l[i0 + lane];
      float len_j = s_r[j0 + lane] - s_l[j0 + lane];
      Am[0] = (_Float16)len_i; Am[1] = (_Float16)1.0f;   // K=0,1 for M=lane
      Bm[0] = (_Float16)1.0f;  Bm[1] = (_Float16)len_j;  // K=0,1 rows for N=lane
    }
    v8f cz = {};
    v8f Dsum = __builtin_amdgcn_wmma_f32_16x16x32_f16(
        false, Am, false, Bm, (short)0, cz, false, false);
    const int j = j0 + (lane & 15);
    const float lj = s_l[j], rj = s_r[j];
#pragma unroll
    for (int r = 0; r < 8; ++r) {
      const int i = i0 + r + ((lane >> 4) << 3);   // lanes 16-31 hold rows M+8
      float inter = fminf(s_r[i], rj) - fmaxf(s_l[i], lj);
      inter = fmaxf(inter, 0.0f);
      if (i < TOPK && j < TOPK &&
          inter * (1.0f + F_OVERLAP) > F_OVERLAP * Dsum[r]) {
        atomicOr(&s_sup[i][j >> 5], 1u << (j & 31));
      }
    }
  }
  __syncthreads();

  // Greedy pass is now pure bit ops on precomputed suppression rows.
  if (tid == 0) {
    unsigned keepw[7] = {0u, 0u, 0u, 0u, 0u, 0u, 0u};
    for (int i = 0; i < TOPK; ++i) {
      const int wi = i >> 5;
      bool sup = false;
      for (int w = 0; w <= wi; ++w) {
        unsigned m = keepw[w] & s_sup[i][w];
        if (w == wi) m &= (1u << (i & 31)) - 1u;   // j < i only
        sup |= (m != 0u);
      }
      if (s_valid[i] && !sup) keepw[wi] |= 1u << (i & 31);
    }
#pragma unroll
    for (int w = 0; w < 7; ++w) s_keepw[w] = keepw[w];
  }
  __syncthreads();
  for (int j = tid; j < TOPK; j += THREADS)
    out_keep[(size_t)bc * TOPK + j] =
        ((s_keepw[j >> 5] >> (j & 31)) & 1u) ? 1.0f : 0.0f;
}

// ---------------- launch ----------------
extern "C" void kernel_launch(void* const* d_in, const int* in_sizes, int n_in,
                              void* d_out, int out_size, void* d_ws, size_t ws_size,
                              hipStream_t stream) {
  const float* loc  = (const float*)d_in[0];   // [16][262144][2]
  const float* cls  = (const float*)d_in[1];   // [16][262144][5]
  // d_in[2] = detection_threshold (unused by the reference computation)
  const float* defs = (const float*)d_in[3];   // [262144][2]
  float* out = (float*)d_out;

  // workspace layout
  unsigned char* ws = (unsigned char*)d_ws;
  unsigned int* hist      = (unsigned int*)ws;                        // 64*1024 u32
  unsigned int* thrBin    = hist + (size_t)NB * CMX * NBINS;          // 64 u32
  unsigned int* candCnt   = thrBin + NB * CMX;                        // 64 u32
  float*        candScore = (float*)(candCnt + NB * CMX);             // 64*2048 f32
  unsigned int* candIdx   = (unsigned int*)(candScore + (size_t)NB * CMX * CAP);

  const int zero_words = NB * CMX * NBINS + NB * CMX + NB * CMX;      // hist+thr+cnt
  zero_ws_kernel<<<(zero_words + 255) / 256, 256, 0, stream>>>(hist, zero_words);

  const int scan_blocks = NB * CHUNKS_PER_B;   // 512
  scan_hist_kernel<<<scan_blocks, THREADS, 0, stream>>>(cls, hist);
  select_thr_kernel<<<1, 64, 0, stream>>>(hist, thrBin);
  compact_kernel<<<scan_blocks, THREADS, 0, stream>>>(cls, thrBin, candCnt,
                                                      candScore, candIdx);
  topk_nms_kernel<<<NB * CMX, THREADS, 0, stream>>>(loc, cls, defs,
                                                    candScore, candIdx, candCnt, out);
}
